// GCN_18459769439026
// MI455X (gfx1250) — matrix operand order repaired and checked by reading, not compiled, not run on hardware
//
#include <hip/hip_runtime.h>

// -------- problem constants (match reference) --------
#define N_NODES 100000
#define N_EDGES 1600000
#define EPS_BN  1e-5f

// -------- WMMA vector types (CDNA5 wave32) --------
typedef __attribute__((ext_vector_type(16))) __bf16 v16bf;
typedef __attribute__((ext_vector_type(8)))  float  v8f;

union FragBF {            // 16 bf16 = 8 dwords per lane
  unsigned int w[8];
  v16bf v;
};

__device__ __forceinline__ unsigned short f2bf(float f) {
  // round-to-nearest-even f32 -> bf16 (bit pattern)
  unsigned int u = __float_as_uint(f);
  unsigned int r = (u + 0x7FFFu + ((u >> 16) & 1u)) >> 16;
  return (unsigned short)r;
}
__device__ __forceinline__ unsigned int pack2(float a, float b) {
  return (unsigned int)f2bf(a) | ((unsigned int)f2bf(b) << 16);
}

// -------- weight pre-transpose + bf16 convert --------
// W1[256][128] -> w1t[n][k] as packed uint pairs: w1t_u32[n*128 + k/2]
__global__ void conv_w1(const float* __restrict__ W1, unsigned int* __restrict__ w1t) {
  int idx = blockIdx.x * 256 + threadIdx.x;      // 0..16383
  int n  = idx >> 7;                              // 0..127
  int k2 = idx & 127;                             // K pair
  w1t[idx] = pack2(W1[(size_t)(2 * k2) * 128 + n], W1[(size_t)(2 * k2 + 1) * 128 + n]);
}
// W2[128][64] -> w2t_u32[n*64 + k/2]
__global__ void conv_w2(const float* __restrict__ W2, unsigned int* __restrict__ w2t) {
  int idx = blockIdx.x * 256 + threadIdx.x;      // 0..4095
  int n  = idx >> 6;                              // 0..63
  int k2 = idx & 63;
  w2t[idx] = pack2(W2[(size_t)(2 * k2) * 64 + n], W2[(size_t)(2 * k2 + 1) * 64 + n]);
}

// -------- degree / symmetric norm --------
__global__ void deg_init(float* __restrict__ dis) {
  int i = blockIdx.x * 256 + threadIdx.x;
  if (i < N_NODES) dis[i] = 1.0f;                // self-loop
}
__global__ void deg_edges(const int* __restrict__ dst, float* __restrict__ dis) {
  int e = blockIdx.x * 256 + threadIdx.x;
  if (e < N_EDGES) atomicAdd(&dis[dst[e]], 1.0f);
}
__global__ void deg_rsqrt(float* __restrict__ dis) {
  int i = blockIdx.x * 256 + threadIdx.x;
  if (i < N_NODES) dis[i] = rsqrtf(dis[i]);      // deg >= 1
}

// -------- GEMM1: h0[N,128] = bf16(x[N,256]) @ bf16(W1) (f32 acc, WMMA) --------
__global__ __launch_bounds__(256) void gemm1_wmma(const float* __restrict__ x,
                                                  const unsigned int* __restrict__ w1t,
                                                  float* __restrict__ h0) {
  __shared__ unsigned int As[16][129];           // 16 rows x 256 K (bf16 pairs), padded
  const int tid = threadIdx.x;
  const int rowBase = blockIdx.x * 16;
  // stage + convert the A tile (coalesced float2 reads)
  #pragma unroll
  for (int i = 0; i < 8; ++i) {
    int u  = tid + i * 256;                      // 0..2047
    int r  = u >> 7, c2 = u & 127;
    const float* p = x + (size_t)(rowBase + r) * 256 + c2 * 2;
    As[r][c2] = pack2(p[0], p[1]);
  }
  __syncthreads();

  const int lane = tid & 31;
  const int wave = tid >> 5;
  const int colBase = wave * 16;                 // 8 waves cover 128 cols
  const int m    = lane & 15;                    // A row / B col / D col
  const int half = lane >> 4;
  __builtin_prefetch(w1t + (size_t)(colBase + m) * 128, 0, 0);

  v8f acc = {};
  #pragma unroll
  for (int kc = 0; kc < 8; ++kc) {               // 8 chunks of K=32
    FragBF a, b;
    #pragma unroll
    for (int t = 0; t < 8; ++t) {                // A-layout: V0..7 -> K 0..15 (+16*half), contiguous
      int ua = t + 8 * half;
      a.w[t] = As[m][kc * 16 + ua];
    }
    const unsigned int* bp = w1t + (size_t)(colBase + m) * 128 + kc * 16 + half * 8;
    #pragma unroll
    for (int t = 0; t < 8; ++t) b.w[t] = bp[t];  // B: lane=N, half picks K 0..15 / 16..31
    acc = __builtin_amdgcn_wmma_f32_16x16x32_bf16(false, a.v, false, b.v,
                                                  (short)0, acc, false, false);
  }
  float* out = h0 + (size_t)rowBase * 128 + colBase;
  #pragma unroll
  for (int r = 0; r < 8; ++r)                    // D: VGPR r -> row r + 8*half, col = m
    out[(size_t)(r + 8 * half) * 128 + m] = acc[r];
}

// -------- GEMM2: z0[N,64] = h1_bf16[N,128] @ bf16(W2) --------
__global__ __launch_bounds__(256) void gemm2_wmma(const unsigned int* __restrict__ h1u,
                                                  const unsigned int* __restrict__ w2t,
                                                  float* __restrict__ z0) {
  __shared__ unsigned int As[32][65];            // 32 rows x 128 K (bf16 pairs), padded
  const int tid = threadIdx.x;
  const int rowBase0 = blockIdx.x * 32;
  #pragma unroll
  for (int i = 0; i < 8; ++i) {
    int u = tid + i * 256;                       // 0..2047
    int r = u >> 6, c2 = u & 63;
    As[r][c2] = h1u[(size_t)(rowBase0 + r) * 64 + c2];
  }
  __syncthreads();

  const int lane = tid & 31;
  const int wave = tid >> 5;
  const int rowT = wave >> 2;                    // 2 row tiles
  const int colBase = (wave & 3) * 16;           // 4 col tiles cover 64 cols
  const int m    = lane & 15;
  const int half = lane >> 4;

  v8f acc = {};
  #pragma unroll
  for (int kc = 0; kc < 4; ++kc) {               // 4 chunks of K=32
    FragBF a, b;
    #pragma unroll
    for (int t = 0; t < 8; ++t) {
      int ua = t + 8 * half;
      a.w[t] = As[rowT * 16 + m][kc * 16 + ua];
    }
    const unsigned int* bp = w2t + (size_t)(colBase + m) * 64 + kc * 16 + half * 8;
    #pragma unroll
    for (int t = 0; t < 8; ++t) b.w[t] = bp[t];
    acc = __builtin_amdgcn_wmma_f32_16x16x32_bf16(false, a.v, false, b.v,
                                                  (short)0, acc, false, false);
  }
  float* out = z0 + (size_t)(rowBase0 + rowT * 16) * 64 + colBase;
  #pragma unroll
  for (int r = 0; r < 8; ++r)
    out[(size_t)(r + 8 * half) * 64 + m] = acc[r];
}

// -------- aggregation: self-loop init, then edge scatter-add (L2-resident) --------
__global__ void agg_init(const float* __restrict__ h, const float* __restrict__ dis,
                         const float* __restrict__ bias, float* __restrict__ out,
                         int cshift) {
  int gid = blockIdx.x * 256 + threadIdx.x;      // < N * cols
  int i = gid >> cshift;
  int c = gid & ((1 << cshift) - 1);
  float dv = dis[i];
  out[gid] = h[gid] * dv * dv + bias[c];
}
__global__ void agg_edges(const int* __restrict__ src, const int* __restrict__ dst,
                          const float* __restrict__ dis, const float* __restrict__ h,
                          float* __restrict__ out, int cols) {
  int gid = blockIdx.x * 256 + threadIdx.x;      // one wave per edge
  int e = gid >> 5;
  if (e >= N_EDGES) return;
  int lane = gid & 31;
  int s = src[e], d = dst[e];
  float coef = dis[s] * dis[d];
  const float* hs = h + (size_t)s * cols;
  float* od = out + (size_t)d * cols;
  for (int c = lane; c < cols; c += 32)
    atomicAdd(&od[c], hs[c] * coef);
}

// -------- batchnorm (training-mode batch stats) + ReLU, emit bf16 for GEMM2 --------
__global__ void zero_stats(float* __restrict__ g) { g[threadIdx.x] = 0.f; } // 256 floats: sum+sumsq
__global__ __launch_bounds__(256) void bn_stats(const float* __restrict__ h,
                                                float* __restrict__ gsum,
                                                float* __restrict__ gsq) {
  __shared__ float ssum[256], ssq[256];
  int c  = threadIdx.x & 127;
  int rh = threadIdx.x >> 7;                     // 2 rows per block per iter
  float s = 0.f, q = 0.f;
  for (int r = blockIdx.x * 2 + rh; r < N_NODES; r += gridDim.x * 2) {
    float v = h[(size_t)r * 128 + c];
    s += v; q += v * v;
  }
  ssum[threadIdx.x] = s; ssq[threadIdx.x] = q;
  __syncthreads();
  if (threadIdx.x < 128) {
    atomicAdd(&gsum[c], ssum[threadIdx.x] + ssum[threadIdx.x + 128]);
    atomicAdd(&gsq[c],  ssq[threadIdx.x]  + ssq[threadIdx.x + 128]);
  }
}
__global__ void bn_finalize(const float* __restrict__ gsum, const float* __restrict__ gsq,
                            const float* __restrict__ gamma, const float* __restrict__ beta,
                            float* __restrict__ scale, float* __restrict__ shift) {
  int c = threadIdx.x;                           // 128 threads
  float mean = gsum[c] * (1.0f / N_NODES);
  float var  = gsq[c]  * (1.0f / N_NODES) - mean * mean;
  float sc = gamma[c] * rsqrtf(var + EPS_BN);
  scale[c] = sc;
  shift[c] = beta[c] - mean * sc;
}
__global__ void bn_apply(const float* __restrict__ h, const float* __restrict__ scale,
                         const float* __restrict__ shift, unsigned short* __restrict__ h1) {
  int gid = blockIdx.x * 256 + threadIdx.x;      // < N*128
  int c = gid & 127;
  float v = h[gid] * scale[c] + shift[c];
  h1[gid] = f2bf(v > 0.f ? v : 0.f);
}

// -------- decode: logits[e] = dot(z[src], z[dst]) --------
__global__ void decode(const int* __restrict__ src, const int* __restrict__ dst,
                       const float* __restrict__ z, float* __restrict__ out) {
  int e = blockIdx.x * 256 + threadIdx.x;
  if (e >= N_EDGES) return;
  const float4* a = (const float4*)(z + (size_t)src[e] * 64);
  const float4* b = (const float4*)(z + (size_t)dst[e] * 64);
  float acc = 0.f;
  #pragma unroll
  for (int i = 0; i < 16; ++i) {
    float4 va = a[i], vb = b[i];
    acc += va.x * vb.x + va.y * vb.y + va.z * vb.z + va.w * vb.w;
  }
  out[e] = acc;
}

extern "C" void kernel_launch(void* const* d_in, const int* in_sizes, int n_in,
                              void* d_out, int out_size, void* d_ws, size_t ws_size,
                              hipStream_t stream) {
  const float* x      = (const float*)d_in[0];
  const int*   ei     = (const int*)d_in[1];
  const float* W1     = (const float*)d_in[2];
  const float* b1     = (const float*)d_in[3];
  const float* gamma1 = (const float*)d_in[4];
  const float* beta1  = (const float*)d_in[5];
  const float* W2     = (const float*)d_in[6];
  const float* b2     = (const float*)d_in[7];
  const int* src = ei;
  const int* dst = ei + N_EDGES;
  float* logits = (float*)d_out;

  // workspace layout (aliased: bufA = h0 then z0, bufB = out1 then z). Peak ~128.5 MB.
  float*          bufA = (float*)d_ws;                               // N*128 f32
  float*          bufB = bufA + (size_t)N_NODES * 128;               // N*128 f32
  unsigned short* h1   = (unsigned short*)(bufB + (size_t)N_NODES * 128); // N*128 bf16
  float*          dis  = (float*)(h1 + (size_t)N_NODES * 128);       // N f32
  unsigned int*   w1t  = (unsigned int*)(dis + N_NODES);             // 128*128 u32
  unsigned int*   w2t  = w1t + 128 * 128;                            // 64*64 u32
  float*          gsum = (float*)(w2t + 64 * 64);                    // 128
  float*          gsq  = gsum + 128;                                 // 128
  float*          scale = gsq + 128;                                 // 128
  float*          shift = scale + 128;                               // 128

  // weights -> bf16 transposed
  conv_w1<<<64, 256, 0, stream>>>(W1, w1t);
  conv_w2<<<16, 256, 0, stream>>>(W2, w2t);

  // symmetric normalization coefficients
  deg_init<<<(N_NODES + 255) / 256, 256, 0, stream>>>(dis);
  deg_edges<<<(N_EDGES + 255) / 256, 256, 0, stream>>>(dst, dis);
  deg_rsqrt<<<(N_NODES + 255) / 256, 256, 0, stream>>>(dis);

  // layer 1: GEMM (WMMA bf16) -> normalized scatter-add (+bias)
  gemm1_wmma<<<N_NODES / 16, 256, 0, stream>>>(x, w1t, bufA);
  agg_init<<<(N_NODES * 128) / 256, 256, 0, stream>>>(bufA, dis, b1, bufB, 7);
  agg_edges<<<(N_EDGES * 32) / 256, 256, 0, stream>>>(src, dst, dis, bufA, bufB, 128);

  // batchnorm + relu -> bf16 activations
  zero_stats<<<1, 256, 0, stream>>>(gsum);
  bn_stats<<<2048, 256, 0, stream>>>(bufB, gsum, gsq);
  bn_finalize<<<1, 128, 0, stream>>>(gsum, gsq, gamma1, beta1, scale, shift);
  bn_apply<<<(N_NODES * 128) / 256, 256, 0, stream>>>(bufB, scale, shift, h1);

  // layer 2: GEMM (WMMA bf16) -> normalized scatter-add (+bias)
  gemm2_wmma<<<N_NODES / 32, 256, 0, stream>>>((const unsigned int*)h1, w2t, bufA);
  agg_init<<<(N_NODES * 64) / 256, 256, 0, stream>>>(bufA, dis, b2, bufB, 6);
  agg_edges<<<(N_EDGES * 32) / 256, 256, 0, stream>>>(src, dst, dis, bufA, bufB, 64);

  // decode: per-edge dot products
  decode<<<N_EDGES / 256, 256, 0, stream>>>(src, dst, bufB, logits);
}
